// IoULoss_83047487636118
// MI455X (gfx1250) — compile-verified, matrix operand order checked by reading, not run on hardware
//
#include <hip/hip_runtime.h>
#include <math.h>

#define EPS_DEN 1e-9f
#define EPS_IN  1e-6f
#define NPAIRS  131072   // 128 * 1024
#define NBLOCKS 512

typedef __attribute__((ext_vector_type(2)))  float    v2f;
typedef __attribute__((ext_vector_type(8)))  float    v8f;
typedef __attribute__((ext_vector_type(16))) _Float16 v16h;

// Wave-level sum via the matrix pipe. A = ones(16x4) so D[m][n] = colsum_n(B)
// for every m; summing the 16 column sums with 4 xor-shuffles yields
// sum(all B elements) = sum(per-lane partials) regardless of B's lane layout.
__device__ __forceinline__ float wave_sum_wmma(float partial) {
#if __has_builtin(__builtin_amdgcn_wmma_f32_16x16x4_f32)
    v2f a; a[0] = 1.0f;    a[1] = 1.0f;   // A = ones (all 64 slots)
    v2f b; b[0] = partial; b[1] = 0.0f;   // B = 32 partials + 32 zeros
    v8f c = {};
    v8f d = __builtin_amdgcn_wmma_f32_16x16x4_f32(
        false, a, false, b, (short)0, c, false, false);
    float s = d[0];                        // lanes 0-15: row M=0; 16-31: row M=8 (identical)
    s += __shfl_xor(s, 1);
    s += __shfl_xor(s, 2);
    s += __shfl_xor(s, 4);
    s += __shfl_xor(s, 8);
    return s;                              // every lane holds the wave total
#else
    // Probe-confirmed fallback: same ones-trick via f16 WMMA (f32 accumulate).
    v16h a, b;
#pragma unroll
    for (int i = 0; i < 16; ++i) { a[i] = (_Float16)1.0f; b[i] = (_Float16)0.0f; }
    b[0] = (_Float16)partial;
    v8f c = {};
    v8f d = __builtin_amdgcn_wmma_f32_16x16x32_f16(
        false, a, false, b, (short)0, c, false, false);
    float s = d[0];
    s += __shfl_xor(s, 1);
    s += __shfl_xor(s, 2);
    s += __shfl_xor(s, 4);
    s += __shfl_xor(s, 8);
    return s;
#endif
}

__global__ __launch_bounds__(256)
void iou_pair_kernel(const float* __restrict__ gt,
                     const float* __restrict__ det,
                     const int*   __restrict__ sizes,
                     float*       __restrict__ blockSums)
{
    const int tid = threadIdx.x;
    const int idx = blockIdx.x * 256 + tid;      // exactly covers NPAIRS
    const int bb  = idx >> 10;                    // batch index (N = 1024)

    // ---- load boxes (coalesced float4) ----
    const float4* g4 = (const float4*)(gt  + (size_t)idx * 8);
    const float4* t4 = (const float4*)(det + (size_t)idx * 8);
    float4 g0 = g4[0], g1 = g4[1];
    float4 q0 = t4[0], q1 = t4[1];
    const float sh = (float)sizes[bb * 2 + 0];    // y scale = sizes[:,0]
    const float sw = (float)sizes[bb * 2 + 1];    // x scale = sizes[:,1]

    float p1x[4], p1y[4], p2x[4], p2y[4];
    p1x[0] = g0.x * sw; p1y[0] = g0.y * sh;
    p1x[1] = g0.z * sw; p1y[1] = g0.w * sh;
    p1x[2] = g1.x * sw; p1y[2] = g1.y * sh;
    p1x[3] = g1.z * sw; p1y[3] = g1.w * sh;
    p2x[0] = q0.x; p2y[0] = q0.y;
    p2x[1] = q0.z; p2y[1] = q0.w;
    p2x[2] = q1.x; p2y[2] = q1.y;
    p2x[3] = q1.z; p2y[3] = q1.w;

    float ptx[24], pty[24], ang[24];
    bool  msk[24];

    // ---- 16 edge-edge intersections (i outer over p1 edges, j inner) ----
#pragma unroll
    for (int i = 0; i < 4; ++i) {
        const float a0x = p1x[i], a0y = p1y[i];
        const float d1x = p1x[(i + 1) & 3] - a0x;
        const float d1y = p1y[(i + 1) & 3] - a0y;
#pragma unroll
        for (int j = 0; j < 4; ++j) {
            const float b0x = p2x[j], b0y = p2y[j];
            const float d2x = p2x[(j + 1) & 3] - b0x;
            const float d2y = p2y[(j + 1) & 3] - b0y;
            const float rx = b0x - a0x, ry = b0y - a0y;
            const float den = d1x * d2y - d1y * d2x;
            const bool  ok  = fabsf(den) > EPS_DEN;
            const float safe = ok ? den : 1.0f;
            const float t = (rx * d2y - ry * d2x) / safe;
            const float u = (rx * d1y - ry * d1x) / safe;
            const bool val = ok && (t >= 0.0f) && (t <= 1.0f)
                                && (u >= 0.0f) && (u <= 1.0f);
            const int k = i * 4 + j;
            ptx[k] = a0x + t * d1x;
            pty[k] = a0y + t * d1y;
            msk[k] = val;
        }
    }

    // ---- p1 vertices inside p2 (two-sided EPS test) ----
#pragma unroll
    for (int i = 0; i < 4; ++i) {
        bool ge = true, le = true;
#pragma unroll
        for (int j = 0; j < 4; ++j) {
            const float ex = p2x[(j + 1) & 3] - p2x[j];
            const float ey = p2y[(j + 1) & 3] - p2y[j];
            const float dx = p1x[i] - p2x[j];
            const float dy = p1y[i] - p2y[j];
            const float cr = ex * dy - ey * dx;
            ge = ge && (cr >= -EPS_IN);
            le = le && (cr <=  EPS_IN);
        }
        ptx[16 + i] = p1x[i]; pty[16 + i] = p1y[i]; msk[16 + i] = ge || le;
    }
    // ---- p2 vertices inside p1 ----
#pragma unroll
    for (int i = 0; i < 4; ++i) {
        bool ge = true, le = true;
#pragma unroll
        for (int j = 0; j < 4; ++j) {
            const float ex = p1x[(j + 1) & 3] - p1x[j];
            const float ey = p1y[(j + 1) & 3] - p1y[j];
            const float dx = p2x[i] - p1x[j];
            const float dy = p2y[i] - p1y[j];
            const float cr = ex * dy - ey * dx;
            ge = ge && (cr >= -EPS_IN);
            le = le && (cr <=  EPS_IN);
        }
        ptx[20 + i] = p2x[i]; pty[20 + i] = p2y[i]; msk[20 + i] = ge || le;
    }

    // ---- masked centroid & angles ----
    int cnt = 0; float cx = 0.0f, cy = 0.0f;
#pragma unroll
    for (int k = 0; k < 24; ++k) {
        cnt += msk[k] ? 1 : 0;
        cx  += msk[k] ? ptx[k] : 0.0f;
        cy  += msk[k] ? pty[k] : 0.0f;
    }
    const float den_c = (float)(cnt > 1 ? cnt : 1);
    cx = cx / den_c;
    cy = cy / den_c;
#pragma unroll
    for (int k = 0; k < 24; ++k)
        ang[k] = msk[k] ? atan2f(pty[k] - cy, ptx[k] - cx) : 1.0e9f;

    // ---- stable odd-even transposition sort by angle (24 passes) ----
#pragma unroll
    for (int pass = 0; pass < 24; ++pass) {
        const int start = pass & 1;
#pragma unroll
        for (int k = 0; k < 23; ++k) {
            if (((k & 1) == start)) {
                const float a0 = ang[k], a1 = ang[k + 1];
                const bool  sw_ = a0 > a1;              // strict => stable
                ang[k]     = sw_ ? a1 : a0;
                ang[k + 1] = sw_ ? a0 : a1;
                const float x0 = ptx[k], x1 = ptx[k + 1];
                ptx[k]     = sw_ ? x1 : x0;
                ptx[k + 1] = sw_ ? x0 : x1;
                const float y0 = pty[k], y1 = pty[k + 1];
                pty[k]     = sw_ ? y1 : y0;
                pty[k + 1] = sw_ ? y0 : y1;
            }
        }
    }

    // ---- replace invalid (angle sentinel) points with first sorted point ----
    const float fx = ptx[0], fy = pty[0];
#pragma unroll
    for (int k = 0; k < 24; ++k) {
        const bool v = ang[k] < 1.0e8f;   // valid angles are in [-pi, pi]
        ptx[k] = v ? ptx[k] : fx;
        pty[k] = v ? pty[k] : fy;
    }

    // ---- shoelace over 24 points ----
    float ssum = 0.0f;
#pragma unroll
    for (int k = 0; k < 24; ++k) {
        const int kn = (k + 1) % 24;      // constant after unroll
        ssum += ptx[k] * pty[kn] - ptx[kn] * pty[k];
    }
    float inter = 0.5f * fabsf(ssum);
    if (cnt < 3) inter = 0.0f;

    // ---- quad areas & IoU ----
    float a1s = 0.0f, a2s = 0.0f;
#pragma unroll
    for (int i = 0; i < 4; ++i) {
        const int j = (i + 1) & 3;
        a1s += p1x[i] * p1y[j] - p1x[j] * p1y[i];
        a2s += p2x[i] * p2y[j] - p2x[j] * p2y[i];
    }
    const float area1 = 0.5f * fabsf(a1s);
    const float area2 = 0.5f * fabsf(a2s);
    const float iou = inter / (area1 + area2 - inter + EPS_DEN);
    const float partial = 1.0f - iou;

    // ---- deterministic block reduction (WMMA ones-trick per wave) ----
    const float wsum = wave_sum_wmma(partial);   // EXEC all-ones here (no tail)
    __shared__ float lds[8];
    const int wave = tid >> 5, lane = tid & 31;
    if (lane == 0) lds[wave] = wsum;
    __syncthreads();
    if (tid == 0) {
        float s = 0.0f;
#pragma unroll
        for (int w = 0; w < 8; ++w) s += lds[w];
        blockSums[blockIdx.x] = s;
    }
}

__global__ __launch_bounds__(256)
void final_reduce_kernel(const float* __restrict__ blockSums,
                         float*       __restrict__ out)
{
    const int tid = threadIdx.x;
    float s = blockSums[tid] + blockSums[tid + 256];
#pragma unroll
    for (int off = 1; off < 32; off <<= 1) s += __shfl_xor(s, off);
    __shared__ float lds[8];
    if ((tid & 31) == 0) lds[tid >> 5] = s;
    __syncthreads();
    if (tid == 0) {
        float t = 0.0f;
#pragma unroll
        for (int w = 0; w < 8; ++w) t += lds[w];
        out[0] = t * (1.0f / (float)NPAIRS);
    }
}

extern "C" void kernel_launch(void* const* d_in, const int* in_sizes, int n_in,
                              void* d_out, int out_size, void* d_ws, size_t ws_size,
                              hipStream_t stream)
{
    const float* gt    = (const float*)d_in[0];   // gt_boxes  (128,1024,8) f32
    const float* det   = (const float*)d_in[1];   // det_boxes (128,1024,8) f32
    const int*   sizes = (const int*)d_in[2];     // sizes     (128,2)      i32
    float* blockSums   = (float*)d_ws;            // 512 floats of scratch
    float* out         = (float*)d_out;           // scalar f32

    iou_pair_kernel<<<NBLOCKS, 256, 0, stream>>>(gt, det, sizes, blockSums);
    final_reduce_kernel<<<1, 256, 0, stream>>>(blockSums, out);
}
